// AU_Net_3573412790684
// MI455X (gfx1250) — matrix operand order repaired and checked by reading, not compiled, not run on hardware
//
#include <hip/hip_runtime.h>

// CDNA5 / gfx1250: wave32, WMMA 16x16x32 bf16 with f32 accumulation.
typedef __attribute__((ext_vector_type(16))) __bf16 v16bf;
typedef __attribute__((ext_vector_type(8)))  float  v8f;

#define NNODE 1026
#define DX_   4096
#define DG_   4096
#define INS_  8192
#define JH_   2048
#define H0_   4096
#define H1_   2048
#define H2_   1024
#define OUTS_ 512
#define NL_   10
#define E_    32832
#define KTOP_ 128
#define ALPHA_ 0.05f
#define PADM  1088   // >= last M-block reach (17*64) for BM=64; rows >= M zero-filled

// ---------------------------------------------------------------- utilities
__global__ void k_fill(float* __restrict__ p, float v, long total) {
  long stride = (long)gridDim.x * blockDim.x;
  for (long i = (long)blockIdx.x * blockDim.x + threadIdx.x; i < total; i += stride) p[i] = v;
}

// f32 -> bf16 row-wise concat of up to 3 sources into a padded [PADM x Kp]
// staging buffer; rows >= M and cols >= (w1+w2+w3) are zero-filled so the
// GEMM A-tile loads need no bounds checks at all.
__global__ void k_cat_bf16(__bf16* __restrict__ dst, int Kp,
                           const float* __restrict__ s1, int w1,
                           const float* __restrict__ s2, int w2,
                           const float* __restrict__ s3, int w3, int M) {
  int Kt = w1 + w2 + w3;
  long total = (long)PADM * Kp;
  long stride = (long)gridDim.x * blockDim.x;
  for (long idx = (long)blockIdx.x * blockDim.x + threadIdx.x; idx < total; idx += stride) {
    int r = (int)(idx / Kp), c = (int)(idx % Kp);
    float v = 0.0f;
    if (r < M && c < Kt) {
      if (c < w1)           v = s1[(long)r * w1 + c];
      else if (c < w1 + w2) v = s2[(long)r * w2 + (c - w1)];
      else                  v = s3[(long)r * w3 + (c - w1 - w2)];
    }
    dst[idx] = (__bf16)v;
  }
}

// (a+b) -> bf16 into padded staging buffer
__global__ void k_sum2_bf16(__bf16* __restrict__ dst, int Kp,
                            const float* __restrict__ a, const float* __restrict__ b,
                            int M, int Kt) {
  long total = (long)PADM * Kp;
  long stride = (long)gridDim.x * blockDim.x;
  for (long idx = (long)blockIdx.x * blockDim.x + threadIdx.x; idx < total; idx += stride) {
    int r = (int)(idx / Kp), c = (int)(idx % Kp);
    float v = 0.0f;
    if (r < M && c < Kt) { long s = (long)r * Kt + c; v = a[s] + b[s]; }
    dst[idx] = (__bf16)v;
  }
}

// ---------------------------------------------------------------- WMMA GEMM
// C[M,N] = act(A_bf16[PADM,lda] @ B_f32[K,N] + bias), f32 accumulate.
// Block tile 64(M) x 128(N), 8 waves in 2x4; each wave computes a 2x2 grid of
// 16x16 WMMA tiles (4 v_wmma per K-chunk, A/B fragments reused).
#define BM 64
#define BN 128
#define KC 32
#define LDP 40  // padded LDS row (halfs): 80B = 20 words, conflict-free, 16B-aligned rows

__global__ __launch_bounds__(256) void k_gemm_bf16(
    const __bf16* __restrict__ A, int lda,
    const float* __restrict__ B,
    const float* __restrict__ bias, float* __restrict__ C,
    int M, int N, int K, int relu)
{
  __shared__ alignas(16) __bf16 As[BM][LDP];   // [m][k]
  __shared__ alignas(16) __bf16 Bs[BN][LDP];   // [n][k] (transposed)
  const int tid  = threadIdx.x;
  const int m0   = blockIdx.y * BM, n0 = blockIdx.x * BN;
  const int wave = tid >> 5, lane = tid & 31;
  const int wr   = wave >> 2, wc = wave & 3;   // 2 x 4 wave grid
  const int half = lane >> 4, ml = lane & 15;
  v8f acc[2][2] = {};
  const int nK = (K + KC - 1) / KC;
  for (int kb = 0; kb < nK; ++kb) {
    const int k0 = kb * KC;
    // ---- A tile: 64x32 bf16, one unguarded 16B load/thread (buffer is padded)
    {
      int r = tid >> 2, c = (tid & 3) * 8;              // 8 contiguous bf16
      const uint4* src = (const uint4*)(A + (long)(m0 + r) * lda + k0 + c);
      *(uint4*)&As[r][c] = *src;
    }
    // ---- B tile: 32(K) x 128(N) f32 -> bf16 transposed
    if (k0 + KC <= K && n0 + BN <= N) {                 // uniform fast path
#pragma unroll
      for (int q = 0; q < 8; ++q) {
        int l = tid + q * 256;                          // float2 index
        int r = l >> 6, c2 = (l & 63) * 2;
        float2 v = *(const float2*)(B + (long)(k0 + r) * N + n0 + c2);
        Bs[c2][r]     = (__bf16)v.x;
        Bs[c2 + 1][r] = (__bf16)v.y;
      }
    } else {                                            // edge tiles: guarded
#pragma unroll
      for (int q = 0; q < 16; ++q) {
        int l = tid + q * 256;
        int r = l >> 7, c = l & 127;
        int gk = k0 + r, gn = n0 + c;
        float v = (gk < K && gn < N) ? B[(long)gk * N + gn] : 0.0f;
        Bs[c][r] = (__bf16)v;
      }
    }
    if (tid == 0 && k0 + KC < K) __builtin_prefetch(&B[(long)(k0 + KC) * N + n0], 0, 1);
    __syncthreads();
    // ---- fragments per CDNA5 ISA wave32 layouts
    v16bf af[2], bfr[2];
#pragma unroll
    for (int t = 0; t < 2; ++t) {
      const int row = wr * 32 + t * 16 + ml;            // A: M = lane&15 (+tile)
#pragma unroll
      for (int i = 0; i < 16; ++i) {                    // K = (i>>3)*16 + half*8 + (i&7)
        int kk = ((i >> 3) << 4) + (half << 3) + (i & 7);
        af[t][i] = As[row][kk];
      }
    }
#pragma unroll
    for (int u = 0; u < 2; ++u) {
      const int col = wc * 16 + u * 64 + ml;            // B: N = lane&15 (+tile)
#pragma unroll
      for (int i = 0; i < 16; ++i)                      // K = (i>>3)*16 + half*8 + (i&7)
        bfr[u][i] = Bs[col][((i >> 3) << 4) + (half << 3) + (i & 7)];
    }
#pragma unroll
    for (int t = 0; t < 2; ++t)
#pragma unroll
      for (int u = 0; u < 2; ++u)
        acc[t][u] = __builtin_amdgcn_wmma_f32_16x16x32_bf16(
            false, af[t], false, bfr[u], (short)0, acc[t][u], false, false);
    __syncthreads();
  }
  // ---- epilogue; D layout: lane col = lane&15, VGPR r -> M = half*8 + r
#pragma unroll
  for (int u = 0; u < 2; ++u) {
    const int nOut = n0 + wc * 16 + u * 64 + ml;
    if (nOut < N) {
      const float bv = bias ? bias[nOut] : 0.0f;
#pragma unroll
      for (int t = 0; t < 2; ++t) {
#pragma unroll
        for (int r = 0; r < 8; ++r) {
          int mOut = m0 + wr * 32 + t * 16 + half * 8 + r;
          if (mOut < M) {
            float v = acc[t][u][r] + bv;
            if (relu) v = v > 0.0f ? v : 0.0f;
            C[(long)mOut * N + nOut] = v;
          }
        }
      }
    }
  }
}

// ---------------------------------------------------------------- GDC (f32)
__global__ void k_add_eye(float* A, int n) {
  int i = blockIdx.x * blockDim.x + threadIdx.x;
  if (i < n) A[(long)i * n + i] += 1.0f;
}

__global__ __launch_bounds__(256) void k_row_dinv(const float* __restrict__ A,
                                                  float* __restrict__ dinv, int n) {
  __shared__ float red[8];
  int i = blockIdx.x;
  float s = 0.0f;
  for (int j = threadIdx.x; j < n; j += 256) s += A[(long)i * n + j];
#pragma unroll
  for (int off = 16; off; off >>= 1) s += __shfl_xor(s, off);
  int wid = threadIdx.x >> 5, lane = threadIdx.x & 31;
  if (lane == 0) red[wid] = s;
  __syncthreads();
  if (threadIdx.x == 0) {
    float t = 0.0f;
    for (int w = 0; w < 8; ++w) t += red[w];
    dinv[i] = rsqrtf(t > 0.0f ? t : 1.0f);
  }
}

// G = [ I - 0.95 * D^-1/2 (A+I) D^-1/2  |  I ],  size n x 2n
__global__ void k_build_aug(float* __restrict__ G, const float* __restrict__ A,
                            const float* __restrict__ dinv, int n) {
  long total = (long)n * 2 * n;
  long stride = (long)gridDim.x * blockDim.x;
  for (long idx = (long)blockIdx.x * blockDim.x + threadIdx.x; idx < total; idx += stride) {
    int i = (int)(idx / (2 * n)), j = (int)(idx % (2 * n));
    float v;
    if (j < n) v = ((i == j) ? 1.0f : 0.0f) -
                   (1.0f - ALPHA_) * dinv[i] * A[(long)i * n + j] * dinv[j];
    else       v = ((j - n) == i) ? 1.0f : 0.0f;
    G[idx] = v;
  }
}

// Snapshot pivot row window [k, n+k] and pivot column (avoids races in elim).
__global__ void k_gj_pivot(const float* __restrict__ G, float* __restrict__ prow,
                           float* __restrict__ pcol, int n, int k) {
  int idx = blockIdx.x * blockDim.x + threadIdx.x;
  int ldg = 2 * n, W = n + 1;
  if (idx < W)          prow[idx] = G[(long)k * ldg + k + idx];
  else if (idx < W + n) pcol[idx - W] = G[(long)(idx - W) * ldg + k];
}

// Gauss-Jordan elimination restricted to active window [k, n+k] (width n+1).
// I - 0.95T is strictly diagonally dominant -> no pivoting needed.
__global__ void k_gj_elim(float* __restrict__ G, const float* __restrict__ prow,
                          const float* __restrict__ pcol, int n, int k) {
  int ldg = 2 * n, W = n + 1;
  long total = (long)n * W;
  long stride = (long)gridDim.x * blockDim.x;
  float pin = 1.0f / prow[0];
  for (long idx = (long)blockIdx.x * blockDim.x + threadIdx.x; idx < total; idx += stride) {
    int i = (int)(idx / W), jj = (int)(idx % W);
    long g = (long)i * ldg + k + jj;
    if (i == k) G[g] = prow[jj] * pin;
    else        G[g] -= pcol[i] * pin * prow[jj];
  }
}

// ST[j][i] = alpha * inv(G)[i][j]  (S transposed: one row per original column)
__global__ void k_extract_ST(float* __restrict__ ST, const float* __restrict__ G,
                             int n, float alpha) {
  long total = (long)n * n;
  long stride = (long)gridDim.x * blockDim.x;
  for (long idx = (long)blockIdx.x * blockDim.x + threadIdx.x; idx < total; idx += stride) {
    int i = (int)(idx / n), j = (int)(idx % n);
    ST[(long)j * n + i] = alpha * G[(long)i * (2 * n) + n + j];
  }
}

// One block per column j (row j of ST): keep exact top-K via repeated argmax
// (first-index tie-break, matching lax.top_k), then column-normalize and
// scatter back to Sn[i][j].
__global__ __launch_bounds__(256) void k_topk_colnorm(const float* __restrict__ ST,
                                                      float* __restrict__ Sn, int n, int kk) {
  __shared__ float vals[1056];
  __shared__ float aux[1056];
  __shared__ unsigned char keep[1056];
  __shared__ float rv[8];
  __shared__ int   ri[8];
  __shared__ float snorm;
  const int j = blockIdx.x, tid = threadIdx.x;
  const int wid = tid >> 5, lane = tid & 31;
  for (int i = tid; i < 1056; i += 256) {
    float v = (i < n) ? ST[(long)j * n + i] : -__builtin_inff();
    vals[i] = v; aux[i] = v; keep[i] = 0;
  }
  __syncthreads();
  for (int t = 0; t < kk; ++t) {
    float bm = -__builtin_inff(); int bi = 0x7fffffff;
    for (int i = tid; i < 1056; i += 256) {
      float v = aux[i];
      if (v > bm) { bm = v; bi = i; }
    }
#pragma unroll
    for (int off = 16; off; off >>= 1) {
      float ov = __shfl_xor(bm, off);
      int   oi = __shfl_xor(bi, off);
      if (ov > bm || (ov == bm && oi < bi)) { bm = ov; bi = oi; }
    }
    if (lane == 0) { rv[wid] = bm; ri[wid] = bi; }
    __syncthreads();
    if (tid == 0) {
      float fm = rv[0]; int fi = ri[0];
      for (int w = 1; w < 8; ++w)
        if (rv[w] > fm || (rv[w] == fm && ri[w] < fi)) { fm = rv[w]; fi = ri[w]; }
      keep[fi] = 1; aux[fi] = -__builtin_inff();
    }
    __syncthreads();
  }
  float s = 0.0f;
  for (int i = tid; i < n; i += 256) if (keep[i]) s += vals[i];
#pragma unroll
  for (int off = 16; off; off >>= 1) s += __shfl_xor(s, off);
  if (lane == 0) rv[wid] = s;
  __syncthreads();
  if (tid == 0) {
    float t = 0.0f;
    for (int w = 0; w < 8; ++w) t += rv[w];
    snorm = (t == 0.0f) ? 1.0f : t;
  }
  __syncthreads();
  float inv = 1.0f / snorm;
  for (int i = tid; i < n; i += 256)
    Sn[(long)i * n + j] = keep[i] ? vals[i] * inv : 0.0f;
}

// ---------------------------------------------------------------- GCN pieces
__global__ void k_deg(const int* __restrict__ ei, float* __restrict__ deg, int E) {
  int e = blockIdx.x * blockDim.x + threadIdx.x;
  if (e < E) atomicAdd(&deg[ei[E + e]], 1.0f);   // identical +1.0 increments: deterministic
}
__global__ void k_rsqrt(float* p, int n) {
  int i = blockIdx.x * blockDim.x + threadIdx.x;
  if (i < n) p[i] = rsqrtf(p[i]);
}
// Dense normalized adjacency: Wadj[c][r] += dinv[r]*dinv[c] per edge.
// All contributions to one cell are the identical value -> order-independent.
__global__ void k_wadj(const int* __restrict__ ei, const float* __restrict__ dinv,
                       float* __restrict__ W, int n, int E) {
  int e = blockIdx.x * blockDim.x + threadIdx.x;
  if (e < E) {
    int r = ei[e], c = ei[E + e];
    atomicAdd(&W[(long)c * n + r], dinv[r] * dinv[c]);
  }
}
__global__ void k_gcn_finish(float* __restrict__ agg, const float* __restrict__ h,
                             const float* __restrict__ dinv, const float* __restrict__ b,
                             int n, int F) {
  long total = (long)n * F;
  long stride = (long)gridDim.x * blockDim.x;
  for (long idx = (long)blockIdx.x * blockDim.x + threadIdx.x; idx < total; idx += stride) {
    int i = (int)(idx / F), f = (int)(idx % F);
    float d = dinv[i];
    float v = agg[idx] + h[idx] * d * d + b[f];
    agg[idx] = v > 0.0f ? v : 0.0f;
  }
}

// ---------------------------------------------------------------- host
static inline int pad64(int k) { return (k + 63) & ~63; }

extern "C" void kernel_launch(void* const* d_in, const int* in_sizes, int n_in,
                              void* d_out, int out_size, void* d_ws, size_t ws_size,
                              hipStream_t stream) {
  (void)in_sizes; (void)n_in; (void)out_size; (void)ws_size;
  const float* x    = (const float*)d_in[0];
  const float* gx   = (const float*)d_in[1];
  const int*   ei   = (const int*)d_in[2];
  const float* jw1  = (const float*)d_in[3];
  const float* jb1  = (const float*)d_in[4];
  const float* jw2  = (const float*)d_in[5];
  const float* jb2  = (const float*)d_in[6];
  const float* ec1w = (const float*)d_in[7];
  const float* ec1b = (const float*)d_in[8];
  const float* drw  = (const float*)d_in[9];
  const float* drb  = (const float*)d_in[10];
  const float* g1w  = (const float*)d_in[11];
  const float* g1b  = (const float*)d_in[12];
  const float* g2w  = (const float*)d_in[13];
  const float* g2b  = (const float*)d_in[14];
  const float* ec2w = (const float*)d_in[15];
  const float* ec2b = (const float*)d_in[16];
  const float* ec3w = (const float*)d_in[17];
  const float* ec3b = (const float*)d_in[18];
  const float* outw = (const float*)d_in[19];
  const float* outb = (const float*)d_in[20];
  float* out = (float*)d_out;

  char* base = (char*)d_ws; size_t off = 0;
  auto alloc = [&](size_t bytes) -> void* {
    void* p = base + off;
    off += (bytes + 255) & ~(size_t)255;
    return p;
  };
  const int n = NNODE;
  __bf16* a_bf = (__bf16*)alloc((size_t)PADM * INS_ * 2);  // padded bf16 A staging
  float* h1   = (float*)alloc((size_t)n * JH_ * 4);
  float* zz   = (float*)alloc((size_t)n * n * 4);
  float* dinN = (float*)alloc((size_t)n * 4);
  float* G    = (float*)alloc((size_t)n * 2 * n * 4);
  float* prow = (float*)alloc((size_t)(n + 1) * 4);
  float* pcol = (float*)alloc((size_t)n * 4);
  float* ST   = (float*)alloc((size_t)n * n * 4);          // later reused as Wadj
  float* Sn   = (float*)alloc((size_t)n * n * 4);
  float* xn   = (float*)alloc((size_t)n * DX_ * 4);
  float* z    = (float*)alloc((size_t)n * H0_ * 4);
  float* z0   = (float*)alloc((size_t)n * H2_ * 4);
  float* hg   = (float*)alloc((size_t)n * H1_ * 4);
  float* z1   = (float*)alloc((size_t)n * H1_ * 4);
  float* z2   = (float*)alloc((size_t)n * H2_ * 4);
  float* zc   = (float*)alloc((size_t)n * H2_ * 4);
  float* zf   = (float*)alloc((size_t)n * OUTS_ * 4);
  float* degv = (float*)alloc((size_t)n * 4);

  auto gsz = [](long total) -> dim3 {
    long b = (total + 255) / 256;
    if (b > 16384) b = 16384;
    return dim3((unsigned)b);
  };
  const int nPK = pad64(n);  // 1088
#define CAT1(dst, Kp, s, w) \
  k_cat_bf16<<<gsz((long)PADM * (Kp)), 256, 0, stream>>>((dst), (Kp), (s), (w), (s), 0, (s), 0, n)
#define GEMM(Abf, ldav, Bp, biasp, Cp, Mv, Nv, Kv, reluv)                         \
  k_gemm_bf16<<<dim3(((Nv) + BN - 1) / BN, ((Mv) + BM - 1) / BM), 256, 0,         \
               stream>>>((Abf), (ldav), (Bp), (biasp), (Cp), (Mv), (Nv), (Kv), (reluv))

  // GCN degree normalization (independent of the rest)
  k_fill<<<gsz(n), 256, 0, stream>>>(degv, 1.0f, n);
  k_deg<<<(E_ + 255) / 256, 256, 0, stream>>>(ei, degv, E_);
  k_rsqrt<<<(n + 255) / 256, 256, 0, stream>>>(degv, n);

  // 1) joint embedding -> dense adjacency zz = relu(relu([x,gx]W1+b1)W2+b2)
  k_cat_bf16<<<gsz((long)PADM * INS_), 256, 0, stream>>>(a_bf, INS_, x, DX_, gx, DG_, gx, 0, n);
  GEMM(a_bf, INS_, jw1, jb1, h1, n, JH_, INS_, 1);
  CAT1(a_bf, JH_, h1, JH_);
  GEMM(a_bf, JH_, jw2, jb2, zz, n, n, JH_, 1);

  // 2) GDC: exact PPR via Gauss-Jordan inverse (f32), topk(128)/col, col-norm
  k_add_eye<<<(n + 255) / 256, 256, 0, stream>>>(zz, n);
  k_row_dinv<<<n, 256, 0, stream>>>(zz, dinN, n);
  k_build_aug<<<gsz((long)n * 2 * n), 256, 0, stream>>>(G, zz, dinN, n);
  for (int k = 0; k < n; ++k) {
    k_gj_pivot<<<(2 * n + 1 + 255) / 256, 256, 0, stream>>>(G, prow, pcol, n, k);
    k_gj_elim<<<gsz((long)n * (n + 1)), 256, 0, stream>>>(G, prow, pcol, n, k);
  }
  k_extract_ST<<<gsz((long)n * n), 256, 0, stream>>>(ST, G, n, ALPHA_);
  k_topk_colnorm<<<n, 256, 0, stream>>>(ST, Sn, n, KTOP_);

  // 3) xn = Sn @ x
  CAT1(a_bf, nPK, Sn, n);
  GEMM(a_bf, nPK, x, (const float*)nullptr, xn, n, DX_, n, 0);

  // 4) z = relu([xn,gx] @ ec1_w + b)
  k_cat_bf16<<<gsz((long)PADM * INS_), 256, 0, stream>>>(a_bf, INS_, xn, DX_, gx, DG_, gx, 0, n);
  GEMM(a_bf, INS_, ec1w, ec1b, z, n, H0_, INS_, 1);

  // 5) z0 = z @ dr_w + b
  CAT1(a_bf, H0_, z, H0_);
  GEMM(a_bf, H0_, drw, drb, z0, n, H2_, H0_, 0);

  // 6) GCN1: h=(z+gx)@g1w ; agg=Wadj@h (dense) ; z1=relu(agg + h*d^2 + b)
  k_sum2_bf16<<<gsz((long)PADM * H0_), 256, 0, stream>>>(a_bf, H0_, z, gx, n, H0_);
  GEMM(a_bf, H0_, g1w, (const float*)nullptr, hg, n, H1_, H0_, 0);
  k_fill<<<gsz((long)n * n), 256, 0, stream>>>(ST, 0.0f, (long)n * n);  // ST := Wadj
  k_wadj<<<(E_ + 255) / 256, 256, 0, stream>>>(ei, degv, ST, n, E_);
  CAT1(a_bf, nPK, ST, n);
  GEMM(a_bf, nPK, hg, (const float*)nullptr, z1, n, H1_, n, 0);
  k_gcn_finish<<<gsz((long)n * H1_), 256, 0, stream>>>(z1, hg, degv, g1b, n, H1_);

  // 7) GCN2
  CAT1(a_bf, H1_, z1, H1_);
  GEMM(a_bf, H1_, g2w, (const float*)nullptr, hg, n, H2_, H1_, 0);
  CAT1(a_bf, nPK, ST, n);
  GEMM(a_bf, nPK, hg, (const float*)nullptr, z2, n, H2_, n, 0);
  k_gcn_finish<<<gsz((long)n * H2_), 256, 0, stream>>>(z2, hg, degv, g2b, n, H2_);

  // 8) zc = relu([z,z1,z2] @ ec2_w + b)
  k_cat_bf16<<<gsz((long)PADM * (H0_ + H1_ + H2_)), 256, 0, stream>>>(
      a_bf, H0_ + H1_ + H2_, z, H0_, z1, H1_, z2, H2_, n);
  GEMM(a_bf, H0_ + H1_ + H2_, ec2w, ec2b, zc, n, H2_, H0_ + H1_ + H2_, 1);

  // 9) zf = relu((zc+z0) @ ec3_w + b)
  k_sum2_bf16<<<gsz((long)PADM * H2_), 256, 0, stream>>>(a_bf, H2_, zc, z0, n, H2_);
  GEMM(a_bf, H2_, ec3w, ec3b, zf, n, OUTS_, H2_, 1);

  // 10) out = zf @ out_w + out_b
  CAT1(a_bf, OUTS_, zf, OUTS_);
  GEMM(a_bf, OUTS_, outw, outb, out, n, NL_, OUTS_, 0);
#undef GEMM
#undef CAT1
}
